// HydraGCN_77687368450216
// MI455X (gfx1250) — matrix operand order repaired and checked by reading, not compiled, not run on hardware
//
#include <hip/hip_runtime.h>
#include <hip/hip_bf16.h>

// ---------------------------------------------------------------------------
// HydraGCN on MI455X (gfx1250): bf16 WMMA flash-attention pipeline.
// Sizes fixed by the reference: b=2, plane=64, n=64*64=4096, heads=4, f=64.
// Round 2: DPP ROW_XMASK softmax reductions (no ds_bpermute), double-buffered
// TDM tile staging to hide tensor-DMA latency behind WMMA work.
// ---------------------------------------------------------------------------

#define B_     2
#define NPIX   4096          // 64*64 pixels per batch
#define CIN    64            // plane
#define HEADS  4
#define FD     64            // inter
#define HF     256           // heads*inter
#define OUTC   64
#define PIX    (B_*NPIX)     // 8192

typedef __attribute__((ext_vector_type(16))) __bf16        v16bf;
typedef __attribute__((ext_vector_type(8)))  float         v8f;
typedef __attribute__((ext_vector_type(4)))  unsigned int  v4u;
typedef __attribute__((ext_vector_type(8)))  int           v8i;
typedef __attribute__((ext_vector_type(4)))  int           v4i;

union Frag16 { unsigned u[8]; v16bf v; };
union Frag8f { float    f[8]; v8f   v; };

#define WMMA_BF16(A,B,C) \
  __builtin_amdgcn_wmma_f32_16x16x32_bf16(false,(A),false,(B),(short)0,(C),false,false)

// ---- cross-lane reductions ------------------------------------------------
// DPP ROW_XMASK: XOR of the 4-bit row-lane id within each 16-lane row --
// exactly the butterfly we need for per-row reductions in the WMMA D layout.
template <int MASK>
__device__ __forceinline__ float dpp_xmask(float x) {
  return __builtin_bit_cast(float,
      __builtin_amdgcn_update_dpp(0, __builtin_bit_cast(int, x),
                                  0x160 + MASK, 0xF, 0xF, true));
}
__device__ __forceinline__ float red16_max(float x) {     // within 16-lane half
  x = fmaxf(x, dpp_xmask<1>(x));
  x = fmaxf(x, dpp_xmask<2>(x));
  x = fmaxf(x, dpp_xmask<4>(x));
  x = fmaxf(x, dpp_xmask<8>(x));
  return x;
}
__device__ __forceinline__ float red16_sum(float x) {
  x += dpp_xmask<1>(x);
  x += dpp_xmask<2>(x);
  x += dpp_xmask<4>(x);
  x += dpp_xmask<8>(x);
  return x;
}
__device__ __forceinline__ float red32_sum(float x) {     // full wave32
#pragma unroll
  for (int m = 1; m < 32; m <<= 1) x += __shfl_xor(x, m, 32);
  return x;
}

// ---- Tensor Data Mover: 2D tile load (D# built per ISA cdna5 ch.8) -------
__device__ __forceinline__ void tdm_load_2d(unsigned lds_addr, unsigned long long gaddr,
                                            unsigned tile0, unsigned tile1,
                                            unsigned td0, unsigned td1, unsigned stride0) {
  v4u g0;
  g0[0] = 1u;                                              // count=1 valid descriptor
  g0[1] = lds_addr;                                        // lds_addr [63:32]
  g0[2] = (unsigned)(gaddr & 0xFFFFFFFFull);               // global_addr lo
  g0[3] = ((unsigned)(gaddr >> 32) & 0x01FFFFFFu) | (2u << 30);  // addr hi + type=2
  v8i g1;
  g1[0] = (int)(1u << 16);                                 // data_size=2B, mask=0
  g1[1] = (int)((td0 & 0xFFFFu) << 16);                    // tensor_dim0 [15:0] @ bits63:48
  g1[2] = (int)((td0 >> 16) | ((td1 & 0xFFFFu) << 16));    // dim0 hi | dim1 lo
  g1[3] = (int)((td1 >> 16) | ((tile0 & 0xFFFFu) << 16));  // dim1 hi | tile_dim0
  g1[4] = (int)(tile1 & 0xFFFFu);                          // tile_dim1, tile_dim2=0
  g1[5] = (int)stride0;                                    // tensor_dim0_stride lo
  g1[6] = 0;
  g1[7] = 0;
  v4i z4 = {0, 0, 0, 0};
#if defined(__clang_major__) && (__clang_major__ >= 23)
  v8i z8 = {0, 0, 0, 0, 0, 0, 0, 0};
  __builtin_amdgcn_tensor_load_to_lds(g0, g1, z4, z4, z8, 0);
#else
  __builtin_amdgcn_tensor_load_to_lds(g0, g1, z4, z4, 0);
#endif
}

// ---------------------------------------------------------------------------
// K0: zero kv + BN stat accumulators (workspace is not re-poisoned to zero)
// ---------------------------------------------------------------------------
__global__ void zero_kernel(float* __restrict__ kv, float* __restrict__ st) {
  int t = threadIdx.x;
  for (int i = t; i < B_ * HF; i += 256) kv[i] = 0.f;
  for (int i = t; i < 512; i += 256) st[i] = 0.f;
}

// ---------------------------------------------------------------------------
// K1: 1x1-conv projections + per-head L2 norm + head means + kv reduction.
// One wave per pixel; lane owns 8 channels (o = lane + 32*i).
// ---------------------------------------------------------------------------
__global__ __launch_bounds__(256) void proj_kernel(
    const float* __restrict__ x,
    const float* __restrict__ wk, const float* __restrict__ bk,
    const float* __restrict__ wq, const float* __restrict__ bq,
    const float* __restrict__ wv, const float* __restrict__ bv,
    __bf16* __restrict__ q_bf, __bf16* __restrict__ km_bf,
    __bf16* __restrict__ qm_bf, float* __restrict__ kv) {
  __shared__ float xs[8][CIN];
  __shared__ float kvloc[HF];
  const int tid = threadIdx.x, wave = tid >> 5, lane = tid & 31;
  const int pixel = blockIdx.x * 8 + wave;
  const int b = pixel / NPIX, n = pixel % NPIX;
  if (tid < HF) kvloc[tid] = 0.f;
  xs[wave][lane]      = x[((size_t)b * CIN + lane) * NPIX + n];
  xs[wave][lane + 32] = x[((size_t)b * CIN + lane + 32) * NPIX + n];
  __syncthreads();

  float ak[8], aq[8], avv[8];
#pragma unroll
  for (int i = 0; i < 8; ++i) {
    int o = lane + 32 * i;
    ak[i] = bk[o]; aq[i] = bq[o]; avv[i] = bv[o];
  }
  for (int f = 0; f < CIN; ++f) {
    float xv = xs[wave][f];
#pragma unroll
    for (int i = 0; i < 8; ++i) {
      int o = lane + 32 * i;
      ak[i]  = fmaf(wk[(size_t)o * CIN + f], xv, ak[i]);
      aq[i]  = fmaf(wq[(size_t)o * CIN + f], xv, aq[i]);
      avv[i] = fmaf(wv[(size_t)o * CIN + f], xv, avv[i]);
    }
  }
  // channel o = 64h + f; i=2h holds f=lane, i=2h+1 holds f=lane+32
  float kn[8], qn[8];
#pragma unroll
  for (int h = 0; h < HEADS; ++h) {
    float ssk = ak[2*h]*ak[2*h] + ak[2*h+1]*ak[2*h+1];
    float ssq = aq[2*h]*aq[2*h] + aq[2*h+1]*aq[2*h+1];
    ssk = red32_sum(ssk); ssq = red32_sum(ssq);
    float ik = 1.f / fmaxf(sqrtf(ssk), 1e-12f);
    float iq = 1.f / fmaxf(sqrtf(ssq), 1e-12f);
    kn[2*h] = ak[2*h]*ik; kn[2*h+1] = ak[2*h+1]*ik;
    qn[2*h] = aq[2*h]*iq; qn[2*h+1] = aq[2*h+1]*iq;
  }
#pragma unroll
  for (int i = 0; i < 8; ++i) {
    int o = lane + 32 * i;
    q_bf[(size_t)pixel * HF + o] = (__bf16)qn[i];
    atomicAdd(&kvloc[o], kn[i] * avv[i]);              // kv[b,h,f] += k*v
  }
  float kml = 0.f, kmh = 0.f, qml = 0.f, qmh = 0.f;
#pragma unroll
  for (int h = 0; h < HEADS; ++h) { kml += kn[2*h]; kmh += kn[2*h+1]; qml += qn[2*h]; qmh += qn[2*h+1]; }
  km_bf[(size_t)pixel * FD + lane]      = (__bf16)(0.25f * kml);
  km_bf[(size_t)pixel * FD + lane + 32] = (__bf16)(0.25f * kmh);
  qm_bf[(size_t)pixel * FD + lane]      = (__bf16)(0.25f * qml);
  qm_bf[(size_t)pixel * FD + lane + 32] = (__bf16)(0.25f * qmh);
  __syncthreads();
  if (tid < HF) atomicAdd(&kv[(size_t)b * HF + tid], kvloc[tid]);
}

// ---------------------------------------------------------------------------
// K2: hydra = q * kv, y1 = hydra @ w1^T + b1, accumulate BN1 stats.
// ---------------------------------------------------------------------------
__global__ __launch_bounds__(256) void h1_kernel(
    const __bf16* __restrict__ q_bf, const float* __restrict__ kv,
    const float* __restrict__ w1, const float* __restrict__ b1,
    float* __restrict__ y1, float* __restrict__ gsum, float* __restrict__ gsq) {
  __shared__ float hyd[8][HF];
  __shared__ float s1[OUTC], s1q[OUTC];
  const int tid = threadIdx.x, wave = tid >> 5, lane = tid & 31;
  const int pixel = blockIdx.x * 8 + wave;
  const int b = pixel / NPIX;
  if (tid < OUTC) { s1[tid] = 0.f; s1q[tid] = 0.f; }
#pragma unroll
  for (int i = 0; i < 8; ++i) {
    int c = lane + 32 * i;
    hyd[wave][c] = (float)q_bf[(size_t)pixel * HF + c] * kv[(size_t)b * HF + c];
  }
  __syncthreads();
#pragma unroll
  for (int r = 0; r < 2; ++r) {
    int o = lane + 32 * r;
    float acc = b1[o];
    for (int c = 0; c < HF; ++c) acc = fmaf(hyd[wave][c], w1[(size_t)o * HF + c], acc);
    y1[(size_t)pixel * OUTC + o] = acc;
    atomicAdd(&s1[o], acc);
    atomicAdd(&s1q[o], acc * acc);
  }
  __syncthreads();
  if (tid < OUTC) { atomicAdd(&gsum[tid], s1[tid]); atomicAdd(&gsq[tid], s1q[tid]); }
}

// ---------------------------------------------------------------------------
// K3: BN stats finalize (mean, rsqrt(var+eps)); biased variance over b*n.
// ---------------------------------------------------------------------------
__global__ void stats_kernel(const float* __restrict__ sum, const float* __restrict__ sq,
                             float* __restrict__ mean, float* __restrict__ istd) {
  int o = threadIdx.x;
  float m = sum[o] * (1.f / (float)PIX);
  float v = sq[o] * (1.f / (float)PIX) - m * m;
  mean[o] = m;
  istd[o] = rsqrtf(v + 1e-5f);
}

// ---------------------------------------------------------------------------
// K4: BN1 + ReLU -> h1, stored bf16 TRANSPOSED [b][out][n] so WMMA B-frag
// K-pairs (adjacent n) are contiguous dwords.
// ---------------------------------------------------------------------------
__global__ void bn1_apply_kernel(const float* __restrict__ y1,
                                 const float* __restrict__ mean, const float* __restrict__ istd,
                                 const float* __restrict__ g, const float* __restrict__ be,
                                 __bf16* __restrict__ h1t) {
  int total = PIX * OUTC;
  for (int idx = blockIdx.x * blockDim.x + threadIdx.x; idx < total; idx += gridDim.x * blockDim.x) {
    int o = idx & (OUTC - 1);
    int pix = idx >> 6;
    int b = pix / NPIX, n = pix % NPIX;
    float v = (y1[idx] - mean[o]) * istd[o] * g[o] + be[o];
    h1t[((size_t)b * OUTC + o) * NPIX + n] = (__bf16)fmaxf(v, 0.f);
  }
}

// ---------------------------------------------------------------------------
// K5: flash attention: softmax(k_mean @ q_mean^T) @ h1, streamed over m.
// 4 waves/WG, each owns a 16-row tile; TDM double-buffers q_mean & h1^T tiles
// into LDS so the DMA for chunk i+1 overlaps the 8 WMMAs of chunk i.
// ---------------------------------------------------------------------------
__global__ __launch_bounds__(128) void attn_kernel(
    const __bf16* __restrict__ km, const __bf16* __restrict__ qm,
    const __bf16* __restrict__ h1t, float* __restrict__ av) {
  __shared__ __bf16 sh_qm[2][32 * FD];     // 2 x (32 m-rows x 64 feat)
  __shared__ __bf16 sh_h1t[2][OUTC * 32];  // 2 x (64 out-cols x 32 m-rows)
  __shared__ __bf16 sh_p[4][16 * 32];      // per-wave P tile (D -> A layout relay)
  const int tid = threadIdx.x, w = tid >> 5, lane = tid & 31;
  const int nt = NPIX / 64;
  const int b = blockIdx.x / nt, tile = blockIdx.x % nt;
  const int r0 = tile * 64 + w * 16;
  const int j16 = lane & 15, kh = lane >> 4;
  const int NITER = NPIX / 32;

  // A fragments: kmean rows r0..r0+15, bf16 16x32 A-layout x2 (K = f 0..63)
  Frag16 a0, a1;
  {
    const unsigned* krow = (const unsigned*)(km + ((size_t)b * NPIX + r0 + j16) * FD);
#pragma unroll
    for (int v = 0; v < 8; ++v) {
      int Kb = 2 * v + 16 * kh;
      a0.u[v] = krow[Kb >> 1];
      a1.u[v] = krow[(Kb >> 1) + 16];
    }
  }

  float mstate[8], lstate[8];
  Frag8f oacc[4];
#pragma unroll
  for (int j = 0; j < 8; ++j) { mstate[j] = -1e30f; lstate[j] = 0.f; }
#pragma unroll
  for (int nc = 0; nc < 4; ++nc)
#pragma unroll
    for (int j = 0; j < 8; ++j) oacc[nc].f[j] = 0.f;

  const unsigned lds_qm0 = (unsigned)(size_t)&sh_qm[0][0];   // LDS offset = low 32 bits
  const unsigned lds_qm1 = (unsigned)(size_t)&sh_qm[1][0];
  const unsigned lds_h10 = (unsigned)(size_t)&sh_h1t[0][0];
  const unsigned lds_h11 = (unsigned)(size_t)&sh_h1t[1][0];
  const unsigned long long gq = (unsigned long long)(size_t)qm  + (size_t)b * NPIX * FD * 2u;
  const unsigned long long gh = (unsigned long long)(size_t)h1t + (size_t)b * OUTC * NPIX * 2u;

  // prologue: stage chunk 0 into buffer 0
  if (w == 0) {
    tdm_load_2d(lds_qm0, gq, FD, 32, FD, NPIX, FD);
    tdm_load_2d(lds_h10, gh, 32, OUTC, NPIX, OUTC, NPIX);
  }

  for (int it = 0; it < NITER; ++it) {
    const int m = it * 32;
    const int buf = it & 1;
    __syncthreads();                       // alternate buffer fully consumed
    if (w == 0) {
      if (it + 1 < NITER) {                // prefetch next chunk into other buffer
        const int mn = m + 32;
        tdm_load_2d(buf ? lds_qm0 : lds_qm1, gq + (size_t)mn * FD * 2u, FD, 32, FD, NPIX, FD);
        tdm_load_2d(buf ? lds_h10 : lds_h11, gh + (size_t)mn * 2u, 32, OUTC, NPIX, OUTC, NPIX);
        __builtin_amdgcn_s_wait_tensorcnt(2);   // current pair done (in-order), next in flight
      } else {
        __builtin_amdgcn_s_wait_tensorcnt(0);
      }
    }
    __syncthreads();                       // current tiles visible to all 4 waves

    // S = kmean_tile(16x64) @ qmean_chunk^T(64x32): 2 tiles x 2 K-chunks
    Frag8f s[2];
#pragma unroll
    for (int t = 0; t < 2; ++t) {
      Frag16 bq0, bq1;
      const unsigned* qb = (const unsigned*)&sh_qm[buf][0] + ((16 * t + j16) * 32 + 8 * kh);
#pragma unroll
      for (int v = 0; v < 8; ++v) { bq0.u[v] = qb[v]; bq1.u[v] = qb[v + 16]; }
      Frag8f cc;
#pragma unroll
      for (int j = 0; j < 8; ++j) cc.f[j] = 0.f;
      cc.v = WMMA_BF16(a0.v, bq0.v, cc.v);
      cc.v = WMMA_BF16(a1.v, bq1.v, cc.v);
      s[t] = cc;
    }

    // online softmax; rows live in 16-lane halves of the D layout, so all
    // reductions are DPP ROW_XMASK butterflies (pure VALU, no LDS traffic).
#pragma unroll
    for (int j = 0; j < 8; ++j) {
      float s0 = s[0].f[j], s1 = s[1].f[j];
      float rm = red16_max(fmaxf(s0, s1));
      float mnew = fmaxf(mstate[j], rm);
      float e0 = __expf(s0 - mnew);
      float e1 = __expf(s1 - mnew);
      float rs = red16_sum(e0 + e1);
      float sc = __expf(mstate[j] - mnew);
      lstate[j] = lstate[j] * sc + rs;
      mstate[j] = mnew;
#pragma unroll
      for (int nc = 0; nc < 4; ++nc) oacc[nc].f[j] *= sc;
      int row = j + 8 * kh;
      sh_p[w][row * 32 + j16]      = (__bf16)e0;
      sh_p[w][row * 32 + 16 + j16] = (__bf16)e1;
    }
    asm volatile("s_wait_dscnt 0" ::: "memory");  // wave-local LDS relay fence

    // P (16x32) as A-fragment
    Frag16 pf;
    {
      const unsigned* pb = (const unsigned*)&sh_p[w][0];
#pragma unroll
      for (int v = 0; v < 8; ++v) {
        int Kb = 2 * v + 16 * kh;
        pf.u[v] = pb[j16 * 16 + (Kb >> 1)];
      }
    }
    // accumulate O += P @ h1_tile (B from transposed LDS tile: contiguous pairs)
#pragma unroll
    for (int nc = 0; nc < 4; ++nc) {
      Frag16 hb;
      const unsigned* hbp = (const unsigned*)&sh_h1t[buf][0] + ((16 * nc + j16) * 16 + 8 * kh);
#pragma unroll
      for (int v = 0; v < 8; ++v) hb.u[v] = hbp[v];
      oacc[nc].v = WMMA_BF16(pf.v, hb.v, oacc[nc].v);
    }
  }

  // epilogue: divide by softmax denominator, write av (f32, [b][n][out])
#pragma unroll
  for (int j = 0; j < 8; ++j) {
    float inv = 1.f / lstate[j];
    int row = r0 + j + 8 * kh;
    float* dst = av + ((size_t)b * NPIX + row) * OUTC;
#pragma unroll
    for (int nc = 0; nc < 4; ++nc) dst[16 * nc + j16] = oacc[nc].f[j] * inv;
  }
}

// ---------------------------------------------------------------------------
// K6: y2 = av @ w2^T + b2, accumulate BN2 stats.
// ---------------------------------------------------------------------------
__global__ __launch_bounds__(256) void y2_kernel(
    const float* __restrict__ av, const float* __restrict__ w2, const float* __restrict__ b2,
    float* __restrict__ y2, float* __restrict__ gsum, float* __restrict__ gsq) {
  __shared__ float avs[8][OUTC];
  __shared__ float s2[OUTC], s2q[OUTC];
  const int tid = threadIdx.x, wave = tid >> 5, lane = tid & 31;
  const int pixel = blockIdx.x * 8 + wave;
  if (tid < OUTC) { s2[tid] = 0.f; s2q[tid] = 0.f; }
  avs[wave][lane]      = av[(size_t)pixel * OUTC + lane];
  avs[wave][lane + 32] = av[(size_t)pixel * OUTC + lane + 32];
  __syncthreads();
#pragma unroll
  for (int r = 0; r < 2; ++r) {
    int o = lane + 32 * r;
    float acc = b2[o];
#pragma unroll
    for (int c = 0; c < OUTC; ++c) acc = fmaf(avs[wave][c], w2[(size_t)o * OUTC + c], acc);
    y2[(size_t)pixel * OUTC + o] = acc;
    atomicAdd(&s2[o], acc);
    atomicAdd(&s2q[o], acc * acc);
  }
  __syncthreads();
  if (tid < OUTC) { atomicAdd(&gsum[tid], s2[tid]); atomicAdd(&gsq[tid], s2q[tid]); }
}

// ---------------------------------------------------------------------------
// K7: BN2 + ReLU + transpose to output layout (b, out, h, w), f32.
// ---------------------------------------------------------------------------
__global__ void out_kernel(const float* __restrict__ y2,
                           const float* __restrict__ mean, const float* __restrict__ istd,
                           const float* __restrict__ g, const float* __restrict__ be,
                           float* __restrict__ out) {
  int total = PIX * OUTC;
  for (int idx = blockIdx.x * blockDim.x + threadIdx.x; idx < total; idx += gridDim.x * blockDim.x) {
    int o = idx & (OUTC - 1);
    int pix = idx >> 6;
    int b = pix / NPIX, n = pix % NPIX;
    float v = (y2[idx] - mean[o]) * istd[o] * g[o] + be[o];
    out[((size_t)b * OUTC + o) * NPIX + n] = fmaxf(v, 0.f);
  }
}

// ---------------------------------------------------------------------------
extern "C" void kernel_launch(void* const* d_in, const int* in_sizes, int n_in,
                              void* d_out, int out_size, void* d_ws, size_t ws_size,
                              hipStream_t stream) {
  const float* x   = (const float*)d_in[0];
  const float* wk  = (const float*)d_in[1];
  const float* bk  = (const float*)d_in[2];
  const float* wq  = (const float*)d_in[3];
  const float* bq  = (const float*)d_in[4];
  const float* wv  = (const float*)d_in[5];
  const float* bv  = (const float*)d_in[6];
  const float* w1  = (const float*)d_in[7];
  const float* b1  = (const float*)d_in[8];
  const float* g1  = (const float*)d_in[9];
  const float* be1 = (const float*)d_in[10];
  const float* w2  = (const float*)d_in[11];
  const float* b2  = (const float*)d_in[12];
  const float* g2  = (const float*)d_in[13];
  const float* be2 = (const float*)d_in[14];
  (void)in_sizes; (void)n_in; (void)out_size; (void)ws_size;

  char* ws = (char*)d_ws;
  size_t off = 0;
  auto alloc = [&](size_t bytes) -> void* {
    void* p = ws + off;
    off = (off + bytes + 255) & ~(size_t)255;
    return p;
  };
  __bf16* q_bf  = (__bf16*)alloc((size_t)PIX * HF * 2);    // 4 MB
  __bf16* km_bf = (__bf16*)alloc((size_t)PIX * FD * 2);    // 1 MB
  __bf16* qm_bf = (__bf16*)alloc((size_t)PIX * FD * 2);    // 1 MB
  __bf16* h1t   = (__bf16*)alloc((size_t)PIX * OUTC * 2);  // 1 MB (transposed)
  float*  kv    = (float*)alloc((size_t)B_ * HF * 4);
  float*  y1    = (float*)alloc((size_t)PIX * OUTC * 4);   // 2 MB
  float*  avb   = (float*)alloc((size_t)PIX * OUTC * 4);   // 2 MB
  float*  y2    = (float*)alloc((size_t)PIX * OUTC * 4);   // 2 MB
  float*  st    = (float*)alloc(512 * 4);  // [sum1 sq1 mean1 istd1 sum2 sq2 mean2 istd2]

  zero_kernel<<<1, 256, 0, stream>>>(kv, st);
  proj_kernel<<<PIX / 8, 256, 0, stream>>>(x, wk, bk, wq, bq, wv, bv, q_bf, km_bf, qm_bf, kv);
  h1_kernel<<<PIX / 8, 256, 0, stream>>>(q_bf, kv, w1, b1, y1, st + 0, st + 64);
  stats_kernel<<<1, 64, 0, stream>>>(st + 0, st + 64, st + 128, st + 192);
  bn1_apply_kernel<<<512, 256, 0, stream>>>(y1, st + 128, st + 192, g1, be1, h1t);
  attn_kernel<<<B_ * (NPIX / 64), 128, 0, stream>>>(km_bf, qm_bf, h1t, avb);
  y2_kernel<<<PIX / 8, 256, 0, stream>>>(avb, w2, b2, y2, st + 256, st + 320);
  stats_kernel<<<1, 64, 0, stream>>>(st + 256, st + 320, st + 384, st + 448);
  out_kernel<<<512, 256, 0, stream>>>(y2, st + 384, st + 448, g2, be2, (float*)d_out);
}
